// LSTMPredictor_33492154974671
// MI455X (gfx1250) — matrix operand order repaired and checked by reading, not compiled, not run on hardware
//
#include <hip/hip_runtime.h>
#include <cstdint>

// ---------------- types ----------------
typedef _Float16 v16h __attribute__((ext_vector_type(16)));
typedef _Float16 h8v  __attribute__((ext_vector_type(8)));
typedef float    v8f  __attribute__((ext_vector_type(8)));
typedef unsigned int u32x4 __attribute__((ext_vector_type(4)));
typedef int i32x4 __attribute__((ext_vector_type(4)));
typedef int i32x8 __attribute__((ext_vector_type(8)));

#define T_STEPS 500
#define BATCH   128
#define HID     512
#define NLAYERS 6
#define GWG     4          // workgroups per layer (each owns 128 h-cols)
#define SMPL    100

// ---------------- helpers ----------------
__device__ __forceinline__ float sigf(float v) { return 1.0f / (1.0f + __expf(-v)); }

__device__ __forceinline__ void wait_ge(int* p, int tgt) {
  while (__hip_atomic_load(p, __ATOMIC_ACQUIRE, __HIP_MEMORY_SCOPE_AGENT) < tgt)
    __builtin_amdgcn_s_sleep(2);
}

// Build an f16 A-fragment (16x32 tile layout): lanes 0-15 hold K 0-7 & 16-23,
// lanes 16-31 hold K 8-15 & 24-31 (p already offset by (lane>>4)*8 halves).
__device__ __forceinline__ v16h make_a16(const _Float16* p) {
  h8v lo = *(const h8v*)p;
  h8v hi = *(const h8v*)(p + 16);
  v16h r;
#pragma unroll
  for (int i = 0; i < 8; ++i) { r[i] = lo[i]; r[i + 8] = hi[i]; }
  return r;
}

// TDM: async DMA a 128x512 f16 row-major tile (stride 512) from global -> LDS.
__device__ __forceinline__ void tdm_load_2d_f16(unsigned lds_off, const void* gaddr) {
  uint64_t ga = (uint64_t)gaddr;
  u32x4 g0;
  g0[0] = 1u;                                             // count=1, user-mode D#
  g0[1] = lds_off;                                        // lds_addr (bytes)
  g0[2] = (unsigned)(ga & 0xFFFFFFFFu);                   // global_addr lo
  g0[3] = (unsigned)((ga >> 32) & 0x1FFFFFFu) | (2u << 30); // addr hi | type=2 (image)
  i32x8 g1;
  g1[0] = (1 << 16);                 // workgroup_mask=0, data_size=1 (2B)
  g1[1] = (int)(512u << 16);         // tensor_dim0 = 512 (bits 63:48)
  g1[2] = (int)(128u << 16);         // dim0 hi=0 | tensor_dim1 = 128
  g1[3] = (int)(512u << 16);         // dim1 hi=0 | tile_dim0 = 512
  g1[4] = 128;                       // tile_dim1 = 128, tile_dim2 = 0
  g1[5] = 512;                       // tensor_dim0_stride lo32
  g1[6] = 0;
  g1[7] = 0;
  i32x4 zz = { 0, 0, 0, 0 };
#if __clang_major__ >= 23
  i32x8 z8 = { 0, 0, 0, 0, 0, 0, 0, 0 };
  __builtin_amdgcn_tensor_load_to_lds(g0, g1, zz, zz, z8, 0);
#else
  __builtin_amdgcn_tensor_load_to_lds(g0, g1, zz, zz, 0);
#endif
}

// ---------------- prologue: weight convert fp32->f16, bias fuse, state zero ----------------
__global__ __launch_bounds__(256) void lstm_prep_k(
    const float* __restrict__ Whh1, const float* __restrict__ WihS, const float* __restrict__ WhhS,
    const float* __restrict__ bih1, const float* __restrict__ bhh1,
    const float* __restrict__ bihS, const float* __restrict__ bhhS,
    _Float16* __restrict__ Whh16, _Float16* __restrict__ Wih16, float* __restrict__ bias,
    float* __restrict__ cstate, _Float16* __restrict__ hbuf, int* __restrict__ done)
{
  int i0 = blockIdx.x * blockDim.x + threadIdx.x;
  int stride = gridDim.x * blockDim.x;
  for (int i = i0; i < 6 * 2048 * 512; i += stride) {
    float v = (i < 2048 * 512) ? Whh1[i] : WhhS[i - 2048 * 512];
    Whh16[i] = (_Float16)v;
  }
  for (int i = i0; i < 5 * 2048 * 512; i += stride) Wih16[i] = (_Float16)WihS[i];
  for (int i = i0; i < 6 * 2048; i += stride)
    bias[i] = (i < 2048) ? (bih1[i] + bhh1[i]) : (bihS[i - 2048] + bhhS[i - 2048]);
  for (int i = i0; i < 6 * 128 * 512; i += stride) cstate[i] = 0.0f;
  for (int i = i0; i < 6 * 2 * 128 * 512; i += stride) hbuf[i] = (_Float16)0.0f;
  for (int i = i0; i < 6; i += stride) done[i] = 0;
}

// ---------------- persistent pipelined LSTM kernel ----------------
// grid = 24 blocks (6 layers x 4 WGs), block = 256 threads (8 waves), dynamic LDS = 256KB
__global__ __launch_bounds__(256) void lstm_persist_k(
    const float* __restrict__ x,        // [500][128][9]
    const float* __restrict__ Wih1,     // [2048][9]
    const _Float16* __restrict__ Whh16, // [6][2048][512]
    const _Float16* __restrict__ Wih16, // [5][2048][512]
    const float* __restrict__ bias,     // [6][2048] (b_ih+b_hh)
    float* __restrict__ cstate,         // [6][128][512]
    _Float16* __restrict__ hbuf,        // [6][2][128][512] ring
    float* __restrict__ hsamp,          // [5][128][512] sampled top-layer h
    int* done)                          // [6] progress counters
{
  extern __shared__ char smem[];
  _Float16* ldsH = (_Float16*)smem;              // 128KB: h_{t-1} of this layer
  _Float16* ldsX = (_Float16*)(smem + 131072);   // 128KB: h_t of layer below
  float*    ldsXf = (float*)(smem + 131072);     // layer0: x[t] (128x9 f32)

  const int tid  = threadIdx.x;
  const int wave = tid >> 5;
  const int lane = tid & 31;
  const int j = blockIdx.x / GWG;     // layer
  const int g = blockIdx.x % GWG;     // column-slice within layer
  const int colbase = g * 128 + wave * 16;       // this wave's 16 h-columns
  const int lcol = colbase + (lane & 15);
  const int kloA = (lane >> 4) * 8;   // A-frag K sub-offset (halves)
  const int kbB  = (lane >> 4) * 8;   // B-frag K sub-offset (halves)

  // biases for this lane's column, all four gates [i,f,g,o]
  const float bI = bias[j * 2048 + 0 * 512 + lcol];
  const float bF = bias[j * 2048 + 1 * 512 + lcol];
  const float bG = bias[j * 2048 + 2 * 512 + lcol];
  const float bO = bias[j * 2048 + 3 * 512 + lcol];

  // layer-0 input weights (K=9) held in registers
  float w1[4][9];
  if (j == 0) {
#pragma unroll
    for (int gt = 0; gt < 4; ++gt)
#pragma unroll
      for (int k = 0; k < 9; ++k) w1[gt][k] = Wih1[(gt * 512 + lcol) * 9 + k];
  }

  const _Float16* WhhL = Whh16 + (size_t)j * 2048 * 512;
  const _Float16* WihL = (j > 0) ? (Wih16 + (size_t)(j - 1) * 2048 * 512) : (const _Float16*)0;
  float* cL = cstate + (size_t)j * 128 * 512;

  const unsigned ldsHoff = (unsigned)(uintptr_t)ldsH;   // flat shared ptr low bits == LDS byte offset
  const unsigned ldsXoff = (unsigned)(uintptr_t)ldsX;

  for (int t = 0; t < T_STEPS; ++t) {
    // ---- cross-workgroup flow control ----
    if (tid == 0) {
      wait_ge(done + j, GWG * t);                            // peers finished t-1
      if (j > 0) wait_ge(done + j - 1, GWG * (t + 1));       // layer below produced h_t
      if (j < NLAYERS - 1 && t >= 2) wait_ge(done + j + 1, GWG * (t - 1)); // consumer drained slot
    }
    __syncthreads();

    // ---- stage activations into LDS via TDM (wave 0 issues) ----
    const _Float16* hprev = hbuf + ((size_t)j * 2 + ((t & 1) ^ 1)) * 128 * 512;
    if (wave == 0) {
      tdm_load_2d_f16(ldsHoff, hprev);
      if (j > 0) {
        const _Float16* xin = hbuf + ((size_t)(j - 1) * 2 + (t & 1)) * 128 * 512;
        tdm_load_2d_f16(ldsXoff, xin);
      }
      __builtin_amdgcn_s_wait_tensorcnt(0);
    }
    if (j == 0) {
      for (int i = tid; i < 128 * 9; i += 256) ldsXf[i] = x[(size_t)t * 128 * 9 + i];
    }
    __syncthreads();

    _Float16* hout = hbuf + ((size_t)j * 2 + (t & 1)) * 128 * 512;
    const bool samp = (j == NLAYERS - 1) && (t % SMPL == SMPL - 1);
    float* hs = hsamp + (size_t)(t / SMPL) * 128 * 512;

    for (int mhalf = 0; mhalf < 2; ++mhalf) {
      v8f acc[4][4];   // [gate][mtile], 16x16 f32 tiles
      v8f vz = { 0, 0, 0, 0, 0, 0, 0, 0 };
#pragma unroll
      for (int gt = 0; gt < 4; ++gt)
#pragma unroll
        for (int mt = 0; mt < 4; ++mt) acc[gt][mt] = vz;

      // ---- recurrent GEMM: A = h_{t-1} (LDS), B = W_hh^T slice (L2) ----
      for (int k0 = 0; k0 < HID; k0 += 32) {
        v16h a[4];
#pragma unroll
        for (int mt = 0; mt < 4; ++mt) {
          const _Float16* ap = ldsH + (size_t)(mhalf * 64 + mt * 16 + (lane & 15)) * 512 + k0 + kloA;
          a[mt] = make_a16(ap);
        }
#pragma unroll
        for (int gt = 0; gt < 4; ++gt) {
          const _Float16* bp = WhhL + (size_t)(gt * 512 + lcol) * 512 + k0 + kbB;
          v16h b = make_a16(bp);
#pragma unroll
          for (int mt = 0; mt < 4; ++mt)
            acc[gt][mt] = __builtin_amdgcn_wmma_f32_16x16x32_f16(
                false, a[mt], false, b, (short)0, acc[gt][mt], false, false);
        }
      }

      if (j > 0) {
        // ---- input GEMM: A = h_t of layer below (LDS), B = W_ih^T slice ----
        for (int k0 = 0; k0 < HID; k0 += 32) {
          v16h a[4];
#pragma unroll
          for (int mt = 0; mt < 4; ++mt) {
            const _Float16* ap = ldsX + (size_t)(mhalf * 64 + mt * 16 + (lane & 15)) * 512 + k0 + kloA;
            a[mt] = make_a16(ap);
          }
#pragma unroll
          for (int gt = 0; gt < 4; ++gt) {
            const _Float16* bp = WihL + (size_t)(gt * 512 + lcol) * 512 + k0 + kbB;
            v16h b = make_a16(bp);
#pragma unroll
            for (int mt = 0; mt < 4; ++mt)
              acc[gt][mt] = __builtin_amdgcn_wmma_f32_16x16x32_f16(
                  false, a[mt], false, b, (short)0, acc[gt][mt], false, false);
          }
        }
      } else {
        // ---- layer-0 input projection, K=9, scalar FMAs from LDS ----
#pragma unroll
        for (int mt = 0; mt < 4; ++mt) {
#pragma unroll
          for (int v = 0; v < 8; ++v) {
            const int m = mhalf * 64 + mt * 16 + (lane >> 4) * 8 + v;
            const float* xr = ldsXf + m * 9;
#pragma unroll
            for (int k = 0; k < 9; ++k) {
              const float xv = xr[k];
              acc[0][mt][v] += xv * w1[0][k];
              acc[1][mt][v] += xv * w1[1][k];
              acc[2][mt][v] += xv * w1[2][k];
              acc[3][mt][v] += xv * w1[3][k];
            }
          }
        }
      }

      // ---- LSTM cell pointwise (all four gate tiles share (lane,vgpr)->(m,col)) ----
#pragma unroll
      for (int mt = 0; mt < 4; ++mt) {
#pragma unroll
        for (int v = 0; v < 8; ++v) {
          const int m = mhalf * 64 + mt * 16 + (lane >> 4) * 8 + v;
          const size_t idx = (size_t)m * 512 + lcol;
          const float iv = acc[0][mt][v] + bI;
          const float fv = acc[1][mt][v] + bF;
          const float gv = acc[2][mt][v] + bG;
          const float ov = acc[3][mt][v] + bO;
          const float cold = cL[idx];
          const float cn = sigf(fv) * cold + sigf(iv) * tanhf(gv);
          const float hn = sigf(ov) * tanhf(cn);
          cL[idx] = cn;
          hout[idx] = (_Float16)hn;
          if (samp) hs[idx] = hn;
        }
      }
    }

    __threadfence();
    __syncthreads();
    if (tid == 0)
      __hip_atomic_fetch_add(done + j, 1, __ATOMIC_RELEASE, __HIP_MEMORY_SCOPE_AGENT);
  }
}

// ---------------- epilogue: 512 -> 3 linear head on the 5 sampled steps ----------------
__global__ __launch_bounds__(256) void lstm_out_k(
    const float* __restrict__ hsamp, const float* __restrict__ Wlin,
    const float* __restrict__ blin, float* __restrict__ out)
{
  int i = blockIdx.x * blockDim.x + threadIdx.x;
  if (i >= 5 * 128 * 3) return;
  const int o = i % 3;
  const int m = (i / 3) % 128;
  const int s = i / (128 * 3);
  const float* h = hsamp + (size_t)s * 128 * 512 + (size_t)m * 512;
  const float* w = Wlin + o * 512;
  float accv = blin[o];
  for (int k = 0; k < 512; ++k) accv += h[k] * w[k];
  out[i] = accv;
}

// ---------------- host launcher ----------------
extern "C" void kernel_launch(void* const* d_in, const int* in_sizes, int n_in,
                              void* d_out, int out_size, void* d_ws, size_t ws_size,
                              hipStream_t stream) {
  const float* x     = (const float*)d_in[0];
  const float* Wih1  = (const float*)d_in[1];
  const float* Whh1  = (const float*)d_in[2];
  const float* bih1  = (const float*)d_in[3];
  const float* bhh1  = (const float*)d_in[4];
  const float* WihS  = (const float*)d_in[5];
  const float* WhhS  = (const float*)d_in[6];
  const float* bihS  = (const float*)d_in[7];
  const float* bhhS  = (const float*)d_in[8];
  const float* Wlin  = (const float*)d_in[9];
  const float* blin  = (const float*)d_in[10];
  float* out = (float*)d_out;

  char* ws = (char*)d_ws;
  size_t oWhh  = 0;
  size_t oWih  = oWhh  + (size_t)6 * 2048 * 512 * 2;
  size_t oBias = oWih  + (size_t)5 * 2048 * 512 * 2;
  size_t oC    = oBias + (size_t)6 * 2048 * 4;
  size_t oHb   = oC    + (size_t)6 * 128 * 512 * 4;
  size_t oHs   = oHb   + (size_t)6 * 2 * 128 * 512 * 2;
  size_t oDn   = oHs   + (size_t)5 * 128 * 512 * 4;
  size_t need  = oDn + 64;
  if (ws_size < need) return;

  _Float16* Whh16 = (_Float16*)(ws + oWhh);
  _Float16* Wih16 = (_Float16*)(ws + oWih);
  float* bias     = (float*)(ws + oBias);
  float* cstate   = (float*)(ws + oC);
  _Float16* hbuf  = (_Float16*)(ws + oHb);
  float* hsamp    = (float*)(ws + oHs);
  int* done       = (int*)(ws + oDn);

  (void)hipFuncSetAttribute(reinterpret_cast<const void*>(lstm_persist_k),
                            hipFuncAttributeMaxDynamicSharedMemorySize, 262144);

  lstm_prep_k<<<1024, 256, 0, stream>>>(Whh1, WihS, WhhS, bih1, bhh1, bihS, bhhS,
                                        Whh16, Wih16, bias, cstate, hbuf, done);

  lstm_persist_k<<<NLAYERS * GWG, 256, 262144, stream>>>(
      x, Wih1, Whh16, Wih16, bias, cstate, hbuf, hsamp, done);

  lstm_out_k<<<(5 * 128 * 3 + 255) / 256, 256, 0, stream>>>(hsamp, Wlin, blin, out);

  (void)in_sizes; (void)n_in; (void)out_size;
}
